// LinearV2E_10290741641935
// MI455X (gfx1250) — compile-verified, matrix operand order checked by reading, not run on hardware
//
#include <hip/hip_runtime.h>

// ---------------------------------------------------------------------------
// Types
// ---------------------------------------------------------------------------
typedef __attribute__((ext_vector_type(16))) __bf16 v16bf;
typedef __attribute__((ext_vector_type(8)))  float  v8f;
typedef __attribute__((ext_vector_type(4)))  unsigned int u32x4;
typedef __attribute__((ext_vector_type(8)))  int    i32x8;
typedef __attribute__((ext_vector_type(4)))  int    i32x4;

#define N_NODES 8192
#define N_EDGES 8192
#define DIM     256
#define HID     256

#if __has_builtin(__builtin_amdgcn_tensor_load_to_lds)
#define HAVE_TDM 1
#else
#define HAVE_TDM 0
#endif

// ---------------------------------------------------------------------------
// Helpers
// ---------------------------------------------------------------------------
__device__ __forceinline__ __bf16 to_bf16(float f) {
  unsigned u = __builtin_bit_cast(unsigned, f);
  u += 0x7FFFu + ((u >> 16) & 1u);            // round-to-nearest-even
  return __builtin_bit_cast(__bf16, (unsigned short)(u >> 16));
}

__device__ __forceinline__ v8f vzero8() {
  v8f z = {0.f, 0.f, 0.f, 0.f, 0.f, 0.f, 0.f, 0.f};
  return z;
}

#if HAVE_TDM
// Issue a 1-D contiguous TDM copy of `nelem` bf16 elements global -> LDS.
// D# layout per CDNA5 ISA. 6-arg builtin form.
__device__ __forceinline__ void tdm_load_1d_bf16(const void* gsrc, void* ldst,
                                                 unsigned nelem) {
  unsigned long long ga = (unsigned long long)(uintptr_t)gsrc;
  unsigned lo = (unsigned)(uintptr_t)ldst;     // LDS offset = low 32 bits of generic addr
  u32x4 g0;
  g0[0] = 1u;                                            // count=1, user mode
  g0[1] = lo;                                            // lds_addr
  g0[2] = (unsigned)(ga & 0xffffffffu);                  // global_addr[31:0]
  g0[3] = (unsigned)((ga >> 32) & 0x1ffffffu) | (2u << 30); // global_addr[56:32] | type=2
  i32x8 g1;
  g1[0] = (int)(1u << 16);                               // workgroup_mask=0, data_size=1 (2B)
  g1[1] = (int)((nelem & 0xffffu) << 16);                // tensor_dim0[15:0]
  g1[2] = (int)(((nelem >> 16) & 0xffffu) | (1u << 16)); // tensor_dim0[31:16] | tensor_dim1=1
  g1[3] = (int)((nelem & 0xffffu) << 16);                // tile_dim0
  g1[4] = (int)1;                                        // tile_dim1=1, tile_dim2=0
  g1[5] = (int)nelem;                                    // tensor_dim0_stride[31:0]
  g1[6] = 0;
  g1[7] = 0;
  i32x4 z4 = {0, 0, 0, 0};
  i32x8 z8 = {0, 0, 0, 0, 0, 0, 0, 0};
  __builtin_amdgcn_tensor_load_to_lds(g0, g1, z4, z4, z8, 0);
}
#endif

// LayerNorm of 16 rows x 256 cols (fp32 in LDS) -> bf16 LDS, blockDim = 128.
__device__ __forceinline__ void layernorm16(const float* lx, __bf16* la,
                                            const float* g, const float* b,
                                            float* red /* 16*16 floats */) {
  const int t = threadIdx.x;          // 0..127
  const int r = t >> 3, s = t & 7;    // row, 1/8th slice
  const float* row = lx + r * 256;
  float sum = 0.f, sq = 0.f;
  #pragma unroll 4
  for (int c = s * 32; c < s * 32 + 32; ++c) { float v = row[c]; sum += v; sq += v * v; }
  red[r * 16 + s]     = sum;
  red[r * 16 + 8 + s] = sq;
  __syncthreads();
  if (t < 16) {
    float S = 0.f, Q = 0.f;
    #pragma unroll
    for (int i = 0; i < 8; ++i) { S += red[t * 16 + i]; Q += red[t * 16 + 8 + i]; }
    float m = S * (1.f / 256.f);
    float v = Q * (1.f / 256.f) - m * m;
    red[t * 16 + 0] = m;
    red[t * 16 + 1] = rsqrtf(v + 1e-5f);
  }
  __syncthreads();
  const float m = red[r * 16 + 0], rs = red[r * 16 + 1];
  #pragma unroll 4
  for (int c = s * 32; c < s * 32 + 32; ++c) {
    la[r * 256 + c] = to_bf16((row[c] - m) * rs * g[c] + b[c]);
  }
  __syncthreads();
}

// One block (128 threads / 4 waves) GEMM: C(16x256) += A(16x256,bf16 LDS row-major)
//  * W(256x256, bf16 global in FRAGMENT-SWIZZLED layout: [(kb*256+col)*32 + kh*16 + j]).
// Weight k-blocks (16 KB contiguous) arrive via TDM, double-buffered against the
// WMMAs of the previous block. Every fragment is a contiguous 32-byte LDS read.
__device__ __forceinline__ void block_gemm_16x256(const __bf16* ldsA,
                                                  const __bf16* gWswz,
                                                  __bf16* ldsW,   // 2 x 32*256 bf16
                                                  v8f acc[4],
                                                  int wave, int lane) {
  const int m = lane & 15, kh = lane >> 4;
  int cur = 0;
#if HAVE_TDM
  if (wave == 0) tdm_load_1d_bf16(gWswz, ldsW, 8192);
#endif
  for (int kb = 0; kb < 8; ++kb) {
#if HAVE_TDM
    if (wave == 0) {
      if (kb + 1 < 8) {
        tdm_load_1d_bf16(gWswz + (kb + 1) * 8192, ldsW + (cur ^ 1) * 8192, 8192);
        __builtin_amdgcn_s_wait_tensorcnt(1);   // older (current) block complete
      } else {
        __builtin_amdgcn_s_wait_tensorcnt(0);
      }
    }
#else
    {
      const uint4* src = (const uint4*)(gWswz + kb * 8192);
      uint4* dst = (uint4*)ldsW;
      uint4 tmp[8];
      #pragma unroll
      for (int u = 0; u < 8; ++u) tmp[u] = src[threadIdx.x + 128 * u];
      #pragma unroll
      for (int u = 0; u < 8; ++u) dst[threadIdx.x + 128 * u] = tmp[u];
    }
#endif
    __syncthreads();
    const __bf16* w = ldsW + cur * 8192;
    v16bf a = *(const v16bf*)(ldsA + m * 256 + kb * 32 + kh * 16);
    #pragma unroll
    for (int t = 0; t < 4; ++t) {
      const int col = wave * 64 + t * 16 + m;
      v16bf b = *(const v16bf*)(w + col * 32 + kh * 16);
      acc[t] = __builtin_amdgcn_wmma_f32_16x16x32_bf16(false, a, false, b,
                                                       (short)0, acc[t], false, false);
    }
    __syncthreads();
#if HAVE_TDM
    cur ^= 1;
#endif
  }
}

// ---------------------------------------------------------------------------
// K0a: fp32 -> bf16 weight conversion INTO fragment-swizzled layout.
// dst[(kb*256+col)*32 + kh*16 + j] = bf16(src[(kb*32+kh*16+j)*256 + col])
// ---------------------------------------------------------------------------
__global__ void cvt_swz_kernel(const float* __restrict__ src, __bf16* __restrict__ dst, int n) {
  int i = blockIdx.x * blockDim.x + threadIdx.x;
  if (i >= n) return;
  const int j   = i & 15;
  const int kh  = (i >> 4) & 1;
  const int col = (i >> 5) & 255;
  const int kb  = i >> 13;
  const int k   = kb * 32 + kh * 16 + j;
  dst[i] = to_bf16(src[k * 256 + col]);
}

// ---------------------------------------------------------------------------
// K0b: fold PE contributions into biases; compute b_all table. One block, 256 thr.
// ---------------------------------------------------------------------------
__global__ void precompute_kernel(const float* __restrict__ m2w1, const float* __restrict__ m2b1,
                                  const float* __restrict__ m3w1, const float* __restrict__ m3b1,
                                  const float* __restrict__ pe2,  const float* __restrict__ pe3,
                                  const float* __restrict__ bpe,  const float* __restrict__ bw1,
                                  const float* __restrict__ bb1,  const float* __restrict__ bw2,
                                  const float* __restrict__ bb2,
                                  float* __restrict__ bias2_row, float* __restrict__ bias2_x0,
                                  float* __restrict__ bias3tab,  float* __restrict__ ball) {
  __shared__ float hid[256];
  const int j = threadIdx.x;
  float s1 = m2b1[j], s0 = m2b1[j];
  for (int k = 0; k < 256; ++k) {
    float w = m2w1[(256 + k) * 256 + j];
    s1 += pe2[256 + k] * w;   // pe2 row 1 (broadcast rows)
    s0 += pe2[k] * w;         // pe2 row 0 (x0 path)
  }
  bias2_row[j] = s1;
  bias2_x0[j]  = s0;
  for (int l = 0; l < 11; ++l) {
    float s = m3b1[j];
    for (int k = 0; k < 256; ++k) s += pe3[l * 256 + k] * m3w1[(256 + k) * 256 + j];
    bias3tab[l * 256 + j] = s;
  }
  for (int l = 0; l < 11; ++l) {
    float h = bb1[j];
    for (int k = 0; k < 128; ++k) h += bpe[l * 128 + k] * bw1[k * 256 + j];
    hid[j] = fmaxf(h, 0.f);
    __syncthreads();
    float o = bb2[j];
    for (int k = 0; k < 256; ++k) o += hid[k] * bw2[k * 256 + j];
    ball[l * 256 + j] = o;
    __syncthreads();
  }
}

// ---------------------------------------------------------------------------
// K1: x <- x + MLP1(LN1(x)); also emit fragment-swizzled bf16(x) and col-sum atomics.
// grid = N/16 blocks, 128 threads. LN output and ReLU hidden share one buffer.
// ---------------------------------------------------------------------------
__global__ void mlp1_kernel(const float* __restrict__ xin,
                            const float* __restrict__ g1, const float* __restrict__ b1,
                            const __bf16* __restrict__ w1, const float* __restrict__ m1b1,
                            const __bf16* __restrict__ w2, const float* __restrict__ m1b2,
                            float* __restrict__ xout, __bf16* __restrict__ xswz,
                            float* __restrict__ colsum) {
  __shared__ __align__(32) float  lx[16 * 256];
  __shared__ __align__(32) __bf16 lact[16 * 256];     // LN out / ReLU hidden (shared)
  __shared__ __align__(32) __bf16 lw[2 * 32 * 256];   // weight staging, double buffer
  __shared__ float red[16 * 16];
  const int tid = threadIdx.x, wave = tid >> 5, lane = tid & 31;
  const size_t row0 = (size_t)blockIdx.x * 16;

  {
    const uint4* srcv = (const uint4*)(xin + row0 * 256);
    uint4* dstv = (uint4*)lx;
    uint4 tmp[8];
    #pragma unroll
    for (int u = 0; u < 8; ++u) tmp[u] = srcv[tid + 128 * u];
    #pragma unroll
    for (int u = 0; u < 8; ++u) dstv[tid + 128 * u] = tmp[u];
  }
  __syncthreads();
  layernorm16(lx, lact, g1, b1, red);

  v8f acc[4];
  #pragma unroll
  for (int t = 0; t < 4; ++t) acc[t] = vzero8();
  block_gemm_16x256(lact, w1, lw, acc, wave, lane);

  // bias + relu -> lact (reads of lact finished inside block_gemm)
  {
    const int nb = wave * 64 + (lane & 15), mb = (lane >> 4) * 8;
    #pragma unroll
    for (int t = 0; t < 4; ++t) {
      const int n = nb + t * 16;
      const float bb = m1b1[n];
      #pragma unroll
      for (int v = 0; v < 8; ++v) {
        float r = acc[t][v] + bb;
        lact[(mb + v) * 256 + n] = to_bf16(r > 0.f ? r : 0.f);
      }
    }
  }
  #pragma unroll
  for (int t = 0; t < 4; ++t) acc[t] = vzero8();
  block_gemm_16x256(lact, w2, lw, acc, wave, lane);

  // residual + store fp32 + swizzled bf16, keep in LDS for column sums
  {
    const int nb = wave * 64 + (lane & 15), mb = (lane >> 4) * 8;
    #pragma unroll
    for (int t = 0; t < 4; ++t) {
      const int n = nb + t * 16;
      const float bb = m1b2[n];
      #pragma unroll
      for (int v = 0; v < 8; ++v) {
        const int m = mb + v;
        float o = lx[m * 256 + n] + acc[t][v] + bb;
        const size_t node = row0 + m;
        xout[node * 256 + n] = o;
        xswz[((node >> 5) * 256 + n) * 32 + (node & 31)] = to_bf16(o);
        lx[m * 256 + n] = o;
      }
    }
  }
  __syncthreads();
  #pragma unroll
  for (int u = 0; u < 2; ++u) {
    const int c = tid + u * 128;
    float s = 0.f;
    #pragma unroll
    for (int r = 0; r < 16; ++r) s += lx[r * 256 + c];
    atomicAdd(&colsum[c], s);
  }
}

// ---------------------------------------------------------------------------
// K2: x0 = mean(x); x0 <- x0 + MLP2(LN2(x0); bias2_x0). One block, 256 threads.
// ---------------------------------------------------------------------------
__global__ void x0_kernel(const float* __restrict__ colsum,
                          const float* __restrict__ n2g, const float* __restrict__ n2b,
                          const float* __restrict__ m2w1, const float* __restrict__ m2w2,
                          const float* __restrict__ m2b2, const float* __restrict__ bias2_x0,
                          float* __restrict__ x0row) {
  __shared__ float xr[256], tn[256], hid[256], red[2];
  const int j = threadIdx.x;
  const float v = colsum[j] * (1.f / (float)N_NODES);
  xr[j] = v;
  __syncthreads();
  if (j == 0) { float s = 0.f; for (int k = 0; k < 256; ++k) s += xr[k]; red[0] = s * (1.f / 256.f); }
  __syncthreads();
  const float m = red[0];
  if (j == 0) { float q = 0.f; for (int k = 0; k < 256; ++k) { float d = xr[k] - m; q += d * d; } red[1] = rsqrtf(q * (1.f / 256.f) + 1e-5f); }
  __syncthreads();
  tn[j] = (v - m) * red[1] * n2g[j] + n2b[j];
  __syncthreads();
  float h = bias2_x0[j];
  for (int k = 0; k < 256; ++k) h += tn[k] * m2w1[k * 256 + j];
  hid[j] = fmaxf(h, 0.f);
  __syncthreads();
  float o = m2b2[j];
  for (int k = 0; k < 256; ++k) o += hid[k] * m2w2[k * 256 + j];
  x0row[j] = v + o;
}

// ---------------------------------------------------------------------------
// K3: x1e = (incidence^T @ x) / prefix_normalizer.  bf16 WMMA.
// grid = E/64 blocks, 256 threads (8 waves). Incidence streams once from HBM;
// swizzled bf16(x) (4 MB) stays L2-resident; x-tiles via double-buffered TDM.
// ---------------------------------------------------------------------------
__global__ void big_gemm_kernel(const float* __restrict__ inc,
                                const __bf16* __restrict__ xswz,
                                const float* __restrict__ norm,
                                float* __restrict__ x1e) {
  __shared__ __align__(32) __bf16 lA[32 * 64];       // incidence^T tile, fragment-major
  __shared__ __align__(32) __bf16 lB[2][32 * 256];   // x tile (swizzled), double buffer
  const int tid = threadIdx.x, wave = tid >> 5, lane = tid & 31;
  const int e0 = blockIdx.x * 64;
  const int eslice = wave >> 1;      // 0..3 : which 16-edge slice
  const int colh   = wave & 1;       // 0..1 : which 128-col half
  const int m = lane & 15, kh = lane >> 4;

  v8f acc[8];
  #pragma unroll
  for (int t = 0; t < 8; ++t) acc[t] = vzero8();

  int cur = 0;
#if HAVE_TDM
  if (wave == 0) tdm_load_1d_bf16(xswz, &lB[0][0], 8192);
#endif

  for (int kb = 0; kb < 256; ++kb) {
    const int n0 = kb * 32;
    if (kb + 1 < 256)
      __builtin_prefetch(inc + (size_t)(n0 + 32) * N_EDGES + e0, 0, 0);  // global_prefetch_b8
    // stage incidence tile (fp32 -> bf16) in fragment-major layout lA[ee*32 + kk]
    {
      float tf[8];
      #pragma unroll
      for (int u = 0; u < 8; ++u) {
        const int i = tid + u * 256;
        const int kk = i >> 6, ee = i & 63;
        tf[u] = inc[(size_t)(n0 + kk) * N_EDGES + e0 + ee];
      }
      #pragma unroll
      for (int u = 0; u < 8; ++u) {
        const int i = tid + u * 256;
        const int kk = i >> 6, ee = i & 63;
        lA[ee * 32 + kk] = to_bf16(tf[u]);
      }
    }
#if HAVE_TDM
    if (wave == 0) {
      if (kb + 1 < 256) {
        tdm_load_1d_bf16(xswz + (size_t)(kb + 1) * 8192, &lB[cur ^ 1][0], 8192);
        __builtin_amdgcn_s_wait_tensorcnt(1);   // older (current) tile complete
      } else {
        __builtin_amdgcn_s_wait_tensorcnt(0);
      }
    }
#else
    {
      const uint4* src = (const uint4*)(xswz + (size_t)kb * 8192);
      uint4* dst = (uint4*)&lB[0][0];
      uint4 tmp[4];
      #pragma unroll
      for (int u = 0; u < 4; ++u) tmp[u] = src[tid + 256 * u];
      #pragma unroll
      for (int u = 0; u < 4; ++u) dst[tid + 256 * u] = tmp[u];
    }
#endif
    __syncthreads();
    const __bf16* lBcur = &lB[cur][0];
    v16bf a = *(const v16bf*)(lA + (eslice * 16 + m) * 32 + kh * 16);
    #pragma unroll
    for (int t = 0; t < 8; ++t) {
      const int col = colh * 128 + t * 16 + m;
      v16bf b = *(const v16bf*)(lBcur + col * 32 + kh * 16);
      acc[t] = __builtin_amdgcn_wmma_f32_16x16x32_bf16(false, a, false, b,
                                                       (short)0, acc[t], false, false);
    }
    __syncthreads();
#if HAVE_TDM
    cur ^= 1;
#endif
  }
  // epilogue: scale by 1/prefix_normalizer[e]
  const int mb = (lane >> 4) * 8;
  #pragma unroll
  for (int t = 0; t < 8; ++t) {
    const int n = colh * 128 + t * 16 + (lane & 15);
    #pragma unroll
    for (int v = 0; v < 8; ++v) {
      const int e = e0 + eslice * 16 + mb + v;
      x1e[(size_t)e * 256 + n] = acc[t][v] * __frcp_rn(norm[e]);
    }
  }
}

// ---------------------------------------------------------------------------
// K4/K5: fused tail. in -> x1 = in + MLP2(LN2(in); bias2_row)
//                     xv = x0 + x1
//                     xv += MLP3(LN3(xv); bias3tab[order])
//                     out = xv + ball[order]
// order == nullptr => use index 1 (vertex path).
// ---------------------------------------------------------------------------
__global__ void tail_kernel(const float* __restrict__ in, const float* __restrict__ x0row,
                            const int* __restrict__ ord,
                            const float* __restrict__ g2, const float* __restrict__ b2,
                            const float* __restrict__ g3, const float* __restrict__ b3,
                            const __bf16* __restrict__ w2a, const float* __restrict__ bias2row,
                            const __bf16* __restrict__ w2b, const float* __restrict__ m2b2,
                            const __bf16* __restrict__ w3a, const float* __restrict__ bias3tab,
                            const __bf16* __restrict__ w3b, const float* __restrict__ m3b2,
                            const float* __restrict__ ball, float* __restrict__ out) {
  __shared__ __align__(32) float  lx[16 * 256];
  __shared__ __align__(32) __bf16 lact[16 * 256];     // LN out / ReLU hidden (shared)
  __shared__ __align__(32) __bf16 lw[2 * 32 * 256];   // weight staging, double buffer
  __shared__ float red[16 * 16];
  const int tid = threadIdx.x, wave = tid >> 5, lane = tid & 31;
  const size_t row0 = (size_t)blockIdx.x * 16;

  {
    const uint4* srcv = (const uint4*)(in + row0 * 256);
    uint4* dstv = (uint4*)lx;
    uint4 tmp[8];
    #pragma unroll
    for (int u = 0; u < 8; ++u) tmp[u] = srcv[tid + 128 * u];
    #pragma unroll
    for (int u = 0; u < 8; ++u) dstv[tid + 128 * u] = tmp[u];
  }
  __syncthreads();
  layernorm16(lx, lact, g2, b2, red);

  v8f acc[4];
  #pragma unroll
  for (int t = 0; t < 4; ++t) acc[t] = vzero8();
  block_gemm_16x256(lact, w2a, lw, acc, wave, lane);
  {
    const int nb = wave * 64 + (lane & 15), mb = (lane >> 4) * 8;
    #pragma unroll
    for (int t = 0; t < 4; ++t) {
      const int n = nb + t * 16;
      const float bb = bias2row[n];
      #pragma unroll
      for (int v = 0; v < 8; ++v) {
        float r = acc[t][v] + bb;
        lact[(mb + v) * 256 + n] = to_bf16(r > 0.f ? r : 0.f);
      }
    }
  }
  #pragma unroll
  for (int t = 0; t < 4; ++t) acc[t] = vzero8();
  block_gemm_16x256(lact, w2b, lw, acc, wave, lane);

  // x1 = in + mlp2 ; xv = x0 + x1  (update lx in place, owner lanes only)
  {
    const int nb = wave * 64 + (lane & 15), mb = (lane >> 4) * 8;
    #pragma unroll
    for (int t = 0; t < 4; ++t) {
      const int n = nb + t * 16;
      const float bb = m2b2[n] + x0row[n];
      #pragma unroll
      for (int v = 0; v < 8; ++v) lx[(mb + v) * 256 + n] += acc[t][v] + bb;
    }
  }
  __syncthreads();
  layernorm16(lx, lact, g3, b3, red);

  #pragma unroll
  for (int t = 0; t < 4; ++t) acc[t] = vzero8();
  block_gemm_16x256(lact, w3a, lw, acc, wave, lane);
  {
    const int nb = wave * 64 + (lane & 15), mb = (lane >> 4) * 8;
    #pragma unroll
    for (int t = 0; t < 4; ++t) {
      const int n = nb + t * 16;
      #pragma unroll
      for (int v = 0; v < 8; ++v) {
        const int m = mb + v;
        int l = 1;
        if (ord) { l = ord[row0 + m]; l = l < 0 ? 0 : (l > 10 ? 10 : l); }
        float r = acc[t][v] + bias3tab[l * 256 + n];
        lact[m * 256 + n] = to_bf16(r > 0.f ? r : 0.f);
      }
    }
  }
  #pragma unroll
  for (int t = 0; t < 4; ++t) acc[t] = vzero8();
  block_gemm_16x256(lact, w3b, lw, acc, wave, lane);
  {
    const int nb = wave * 64 + (lane & 15), mb = (lane >> 4) * 8;
    #pragma unroll
    for (int t = 0; t < 4; ++t) {
      const int n = nb + t * 16;
      const float bb = m3b2[n];
      #pragma unroll
      for (int v = 0; v < 8; ++v) {
        const int m = mb + v;
        int l = 1;
        if (ord) { l = ord[row0 + m]; l = l < 0 ? 0 : (l > 10 ? 10 : l); }
        out[(row0 + m) * 256 + n] = lx[m * 256 + n] + acc[t][v] + bb + ball[l * 256 + n];
      }
    }
  }
}

// ---------------------------------------------------------------------------
// Host launcher
// ---------------------------------------------------------------------------
extern "C" void kernel_launch(void* const* d_in, const int* in_sizes, int n_in,
                              void* d_out, int out_size, void* d_ws, size_t ws_size,
                              hipStream_t stream) {
  (void)in_sizes; (void)n_in; (void)out_size; (void)ws_size;
  const float* x     = (const float*)d_in[0];
  const float* inc   = (const float*)d_in[1];
  const int*   ords  = (const int*)d_in[2];
  const float* norm  = (const float*)d_in[3];
  const float* m1w1  = (const float*)d_in[4];
  const float* m1b1  = (const float*)d_in[5];
  const float* m1w2  = (const float*)d_in[6];
  const float* m1b2  = (const float*)d_in[7];
  const float* m2w1  = (const float*)d_in[8];
  const float* m2b1  = (const float*)d_in[9];
  const float* m2w2  = (const float*)d_in[10];
  const float* m2b2  = (const float*)d_in[11];
  const float* m3w1  = (const float*)d_in[12];
  const float* m3b1  = (const float*)d_in[13];
  const float* m3w2  = (const float*)d_in[14];
  const float* m3b2  = (const float*)d_in[15];
  const float* n1g   = (const float*)d_in[16];
  const float* n1b   = (const float*)d_in[17];
  const float* n2g   = (const float*)d_in[18];
  const float* n2b   = (const float*)d_in[19];
  const float* n3g   = (const float*)d_in[20];
  const float* n3b   = (const float*)d_in[21];
  const float* pe2   = (const float*)d_in[22];
  const float* pe3   = (const float*)d_in[23];
  const float* bpe   = (const float*)d_in[24];
  const float* bw1   = (const float*)d_in[25];
  const float* bb1   = (const float*)d_in[26];
  const float* bw2   = (const float*)d_in[27];
  const float* bb2   = (const float*)d_in[28];

  char* ws = (char*)d_ws;
  size_t off = 0;
  auto alloc = [&](size_t bytes) -> void* {
    void* p = ws + off;
    off += (bytes + 255) & ~(size_t)255;
    return p;
  };
  float*  xbuf      = (float*)alloc((size_t)N_NODES * DIM * 4);
  __bf16* xswz      = (__bf16*)alloc((size_t)N_NODES * DIM * 2);
  float*  x1e       = (float*)alloc((size_t)N_EDGES * DIM * 4);
  float*  colsum    = (float*)alloc(256 * 4);
  float*  x0row     = (float*)alloc(256 * 4);
  float*  bias2_row = (float*)alloc(256 * 4);
  float*  bias2_x0  = (float*)alloc(256 * 4);
  float*  bias3tab  = (float*)alloc(11 * 256 * 4);
  float*  ball      = (float*)alloc(11 * 256 * 4);
  __bf16* m1w1b     = (__bf16*)alloc(65536 * 2);
  __bf16* m1w2b     = (__bf16*)alloc(65536 * 2);
  __bf16* m2w1b     = (__bf16*)alloc(65536 * 2);  // first D rows only
  __bf16* m2w2b     = (__bf16*)alloc(65536 * 2);
  __bf16* m3w1b     = (__bf16*)alloc(65536 * 2);  // first D rows only
  __bf16* m3w2b     = (__bf16*)alloc(65536 * 2);

  (void)hipMemsetAsync(colsum, 0, 256 * 4, stream);

  const int CV_B = 256, CV_G = (65536 + CV_B - 1) / CV_B;
  cvt_swz_kernel<<<CV_G, CV_B, 0, stream>>>(m1w1, m1w1b, 65536);
  cvt_swz_kernel<<<CV_G, CV_B, 0, stream>>>(m1w2, m1w2b, 65536);
  cvt_swz_kernel<<<CV_G, CV_B, 0, stream>>>(m2w1, m2w1b, 65536);  // rows 0..255 of (512,256)
  cvt_swz_kernel<<<CV_G, CV_B, 0, stream>>>(m2w2, m2w2b, 65536);
  cvt_swz_kernel<<<CV_G, CV_B, 0, stream>>>(m3w1, m3w1b, 65536);  // rows 0..255 of (512,256)
  cvt_swz_kernel<<<CV_G, CV_B, 0, stream>>>(m3w2, m3w2b, 65536);

  precompute_kernel<<<1, 256, 0, stream>>>(m2w1, m2b1, m3w1, m3b1, pe2, pe3,
                                           bpe, bw1, bb1, bw2, bb2,
                                           bias2_row, bias2_x0, bias3tab, ball);

  mlp1_kernel<<<N_NODES / 16, 128, 0, stream>>>(x, n1g, n1b, m1w1b, m1b1,
                                                m1w2b, m1b2, xbuf, xswz, colsum);

  x0_kernel<<<1, 256, 0, stream>>>(colsum, n2g, n2b, m2w1, m2w2, m2b2, bias2_x0, x0row);

  big_gemm_kernel<<<N_EDGES / 64, 256, 0, stream>>>(inc, xswz, norm, x1e);

  float* out_v = (float*)d_out;
  float* out_e = out_v + (size_t)N_NODES * DIM;

  tail_kernel<<<N_NODES / 16, 128, 0, stream>>>(xbuf, x0row, nullptr,
                                                n2g, n2b, n3g, n3b,
                                                m2w1b, bias2_row, m2w2b, m2b2,
                                                m3w1b, bias3tab, m3w2b, m3b2,
                                                ball, out_v);

  tail_kernel<<<N_EDGES / 16, 128, 0, stream>>>(x1e, x0row, ords,
                                                n2g, n2b, n3g, n3b,
                                                m2w1b, bias2_row, m2w2b, m2b2,
                                                m3w1b, bias3tab, m3w2b, m3b2,
                                                ball, out_e);
}
